// DRDM_15298673508965
// MI455X (gfx1250) — matrix operand order repaired and checked by reading, not compiled, not run on hardware
//
#include <hip/hip_runtime.h>
#include <hip/hip_bf16.h>
#include <math.h>

#define HH 128
#define WW_ 128
#define HW (HH * WW_)          // 16384
#define CC 64
#define BB 2
#define KK 11
#define LDIM (CC * KK)         // 704
#define FSZ (BB * CC * HW)     // 2097152 elements per feature map
#define INV_N (1.f / 32768.f)  // BN over (B,H,W)
#define EPSV 1e-5f

typedef __attribute__((ext_vector_type(16))) __bf16 v16bf;
typedef __attribute__((ext_vector_type(8)))  float  v8f;

union BF16Frag {
  v16bf v;
  uint4 q[2];
  unsigned short s[16];
};

static __device__ __forceinline__ unsigned short f2bf(float f) {
  unsigned int u = __float_as_uint(f);
  u += 0x7FFFu + ((u >> 16) & 1u);
  return (unsigned short)(u >> 16);
}

// per-channel BN-train affine from (sum,sumsq) accumulators
static __device__ __forceinline__ void bn_affine(const float* acc, const float* g,
                                                 const float* b, int c,
                                                 float& sc, float& sh) {
  float m = acc[2 * c] * INV_N;
  float v = acc[2 * c + 1] * INV_N - m * m;
  sc = rsqrtf(v + EPSV) * g[c];
  sh = b[c] - m * sc;
}

// wave32 reduction of (sum, sumsq) -> one global atomic per wave
static __device__ __forceinline__ void wave_acc(float s, float* acc, int c) {
  float q = s * s;
  for (int o = 16; o > 0; o >>= 1) {
    s += __shfl_down(s, o, 32);
    q += __shfl_down(q, o, 32);
  }
  if ((threadIdx.x & 31) == 0) {
    atomicAdd(&acc[2 * c], s);
    atomicAdd(&acc[2 * c + 1], q);
  }
}

// ---------------------------------------------------------------- bf16 convert
__global__ __launch_bounds__(256) void k_f32_to_bf16(
    const float* __restrict__ src, unsigned short* __restrict__ dst, int n) {
  int i = blockIdx.x * 256 + threadIdx.x;
  if (i < n) dst[i] = f2bf(src[i]);
}

// ------------- depthwise conv, optional inline BN+ReLU on input, + BN reduce
__global__ __launch_bounds__(256) void k_dwconv(
    const float* __restrict__ in, const float* __restrict__ accin,
    const float* __restrict__ gin, const float* __restrict__ bin,
    const float* __restrict__ w, float* __restrict__ out,
    float* __restrict__ accout, int kh, int kw, int normin) {
  int idx = blockIdx.x * 256 + threadIdx.x;
  int x = idx % WW_;
  int y = (idx / WW_) % HH;
  int c = (idx / HW) % CC;   // constant within block (HW % 256 == 0)
  int bc = idx / HW;
  const float* ip = in + (size_t)bc * HW;
  float sc = 1.f, sh = 0.f;
  if (normin) bn_affine(accin, gin, bin, c, sc, sh);
  int oy = kh >> 1, ox = kw >> 1;
  float s = 0.f;
  for (int i = 0; i < kh; ++i) {
    int yy = y + i - oy;
    if (yy < 0 || yy >= HH) continue;
    for (int j = 0; j < kw; ++j) {
      int xx = x + j - ox;
      if (xx < 0 || xx >= WW_) continue;
      float t = ip[yy * WW_ + xx];
      if (normin) t = fmaxf(t * sc + sh, 0.f);
      s += t * w[(c * kh + i) * kw + j];
    }
  }
  out[idx] = s;
  wave_acc(s, accout, c);
}

// ------- offset conv 3x3 (64 -> 22), inline BN+ReLU on feature, + BN reduce
__global__ __launch_bounds__(256) void k_offconv(
    const float* __restrict__ f, const float* __restrict__ accf,
    const float* __restrict__ gf, const float* __restrict__ bf,
    const float* __restrict__ w, const float* __restrict__ bias,
    float* __restrict__ out, float* __restrict__ acc) {
  __shared__ float scf[CC], shf[CC];
  int idx = blockIdx.x * 256 + threadIdx.x;
  if (threadIdx.x < CC) bn_affine(accf, gf, bf, threadIdx.x, scf[threadIdx.x], shf[threadIdx.x]);
  __syncthreads();
  int x = idx % WW_;
  int y = (idx / WW_) % HH;
  int oc = (idx / HW) % (2 * KK);
  int b = idx / (2 * KK * HW);
  float s = bias[oc];
  const float* fb = f + (size_t)b * CC * HW;
  for (int ic = 0; ic < CC; ++ic) {
    const float* fp = fb + (size_t)ic * HW;
    const float* wp = w + (oc * CC + ic) * 9;
    float sci = scf[ic], shi = shf[ic];
    for (int dy = 0; dy < 3; ++dy) {
      int yy = y + dy - 1;
      if (yy < 0 || yy >= HH) continue;
      for (int dx = 0; dx < 3; ++dx) {
        int xx = x + dx - 1;
        if (xx < 0 || xx >= WW_) continue;
        s += fmaxf(fp[yy * WW_ + xx] * sci + shi, 0.f) * wp[dy * 3 + dx];
      }
    }
  }
  out[idx] = s;
  wave_acc(s, acc, oc);
}

// ------------- deformable sampling + bf16 WMMA GEMM (einsum bkhwc,ock->bohw)
// feature BN+ReLU and offset BN+tanh applied inline from accumulators.
__global__ __launch_bounds__(32) void k_dsconv_wmma(
    const float* __restrict__ f, const float* __restrict__ accf,
    const float* __restrict__ gf, const float* __restrict__ bfp,
    const float* __restrict__ offraw, const float* __restrict__ acco,
    const float* __restrict__ go, const float* __restrict__ bo,
    const unsigned short* __restrict__ wbf, const float* __restrict__ dsb,
    float* __restrict__ outds, float* __restrict__ accgn, int morph) {
  const int lane = threadIdx.x;
  const int pt = blockIdx.x * 16;          // pixel tile over B*H*W
  const int b = pt / HW;
  const int pbase = pt % HW;               // tile never crosses batch (HW%16==0)

  __shared__ float scf[CC], shf[CC];
  __shared__ float sco[2 * KK], sho[2 * KK];
  __shared__ float syy[16 * KK];
  __shared__ float sxx[16 * KK];
  __shared__ __align__(32) unsigned short S[16 * LDIM];  // bf16 sampled tile
  __shared__ float gs[16], gss[16];

  // ---- phase 0: per-channel affine constants into LDS
  for (int c = lane; c < CC; c += 32) bn_affine(accf, gf, bfp, c, scf[c], shf[c]);
  if (lane < 2 * KK) bn_affine(acco, go, bo, lane, sco[lane], sho[lane]);
  if (lane < 16) { gs[lane] = 0.f; gss[lane] = 0.f; }
  __syncthreads();

  // ---- phase 1: coordinates with cumulative snake offsets (tanh(bn(raw)))
  for (int idx = lane; idx < 16 * KK; idx += 32) {
    int p = idx / KK, k = idx % KK;
    int pix = pbase + p;
    int h = pix / WW_, w = pix % WW_;
    int cb = morph ? KK : 0;
    const float* ob = offraw + (size_t)b * 2 * KK * HW;
    float cum = 0.f;
    if (k == 0)       cum = tanhf(ob[(size_t)cb * HW + pix] * sco[cb] + sho[cb]);
    else if (k == 10) cum = tanhf(ob[(size_t)(cb + 10) * HW + pix] * sco[cb + 10] + sho[cb + 10]);
    else if (k < 5) {
      for (int t = k; t <= 4; ++t)
        cum += tanhf(ob[(size_t)(cb + t) * HW + pix] * sco[cb + t] + sho[cb + t]);
    } else if (k > 5) {
      for (int t = 6; t <= k; ++t)
        cum += tanhf(ob[(size_t)(cb + t) * HW + pix] * sco[cb + t] + sho[cb + t]);
    }
    float yy, xx;
    if (morph == 0) { yy = (float)h + cum; xx = (float)(w + k - 5); }
    else            { xx = (float)w + cum; yy = (float)(h + k - 5); }
    syy[idx] = yy;
    sxx[idx] = xx;
  }
  __syncthreads();

  // ---- phase 2: bilinear sampling of BN+ReLU'd feature -> bf16 S[p][c*11+k]
  // pixel-fastest decomposition: adjacent lanes hit contiguous addresses.
  for (int sidx = lane; sidx < 16 * LDIM; sidx += 32) {
    int p = sidx & 15;
    int ck = sidx >> 4;
    int c = ck / KK, k = ck - c * KK;
    float yy = syy[p * KK + k], xx = sxx[p * KK + k];
    int y0 = (int)floorf(yy), x0 = (int)floorf(xx);
    int y0i = min(max(y0, 0), HH - 1), y1i = min(max(y0 + 1, 0), HH - 1);
    int x0i = min(max(x0, 0), WW_ - 1), x1i = min(max(x0 + 1, 0), WW_ - 1);
    float y0f = (float)y0i, y1f = (float)y1i, x0f = (float)x0i, x1f = (float)x1i;
    float wa = (y1f - yy) * (x1f - xx);
    float wb = (y1f - yy) * (xx - x0f);
    float wc = (yy - y0f) * (x1f - xx);
    float wd = (yy - y0f) * (xx - x0f);
    const float* fc = f + (size_t)(b * CC + c) * HW;
    float sc = scf[c], sh = shf[c];
    float ta = fmaxf(fc[y0i * WW_ + x0i] * sc + sh, 0.f);
    float tb = fmaxf(fc[y0i * WW_ + x1i] * sc + sh, 0.f);
    float tc = fmaxf(fc[y1i * WW_ + x0i] * sc + sh, 0.f);
    float td = fmaxf(fc[y1i * WW_ + x1i] * sc + sh, 0.f);
    S[p * LDIM + c * KK + k] = f2bf(wa * ta + wb * tb + wc * tc + wd * td);
  }
  __syncthreads();

  // ---- phase 3: D(64x16) = W(64x704) * S^T(704x16), bf16 WMMA, f32 acc
  v8f accv[4];
  const v8f vzero = {0.f, 0.f, 0.f, 0.f, 0.f, 0.f, 0.f, 0.f};
  for (int m = 0; m < 4; ++m) accv[m] = vzero;
  const bool hi = lane >= 16;
  const int lcol = lane & 15;
  for (int ks = 0; ks < LDIM / 32; ++ks) {       // 22 K-steps of 32
    int Kb = ks * 32;
    BF16Frag bf;                                  // B fragment from LDS
    const uint4* bp =
        reinterpret_cast<const uint4*>(&S[lcol * LDIM + Kb + (hi ? 8 : 0)]);
    bf.q[0] = bp[0];
    bf.q[1] = bp[2];
    // prefetch next A-tile line (-> global_prefetch_b8)
    __builtin_prefetch(&wbf[(size_t)lcol * LDIM + Kb + 32], 0, 0);
    for (int m = 0; m < 4; ++m) {
      int o = m * 16 + lcol;                      // A row = out channel
      BF16Frag af;
      const uint4* ap = reinterpret_cast<const uint4*>(
          &wbf[(size_t)o * LDIM + Kb + (hi ? 8 : 0)]);
      af.q[0] = ap[0];      // K[base .. base+7]
      af.q[1] = ap[2];      // K[base+16 .. base+23]
      accv[m] = __builtin_amdgcn_wmma_f32_16x16x32_bf16(
          false, af.v, false, bf.v, (short)0, accv[m], false, false);
    }
  }

  // ---- phase 4: + ds_b, store NCHW, group-norm partial sums (4 ch / group)
  const int rowoff = hi ? 8 : 0;
  for (int m = 0; m < 4; ++m) {
    float ls0 = 0.f, lq0 = 0.f, ls1 = 0.f, lq1 = 0.f;
    for (int r = 0; r < 8; ++r) {
      int o = m * 16 + rowoff + r;
      float v = accv[m][r] + dsb[o];
      outds[(size_t)(b * CC + o) * HW + pbase + lcol] = v;
      if (r < 4) { ls0 += v; lq0 += v * v; }
      else       { ls1 += v; lq1 += v * v; }
    }
    int g0 = (m * 16 + rowoff) >> 2;
    atomicAdd(&gs[g0], ls0);
    atomicAdd(&gss[g0], lq0);
    atomicAdd(&gs[g0 + 1], ls1);
    atomicAdd(&gss[g0 + 1], lq1);
  }
  __syncthreads();
  if (lane < 16) {
    atomicAdd(&accgn[(b * 16 + lane) * 2], gs[lane]);
    atomicAdd(&accgn[(b * 16 + lane) * 2 + 1], gss[lane]);
  }
}

// ---- fuse: x0 = x + gnrelu(ds0) + gnrelu(ds1) + bnrelu(x3raw); BN reduce x0
__global__ __launch_bounds__(256) void k_sum4_reduce(
    const float* __restrict__ x,
    const float* __restrict__ ds0, const float* __restrict__ accG0,
    const float* __restrict__ gg0, const float* __restrict__ gb0,
    const float* __restrict__ ds1, const float* __restrict__ accG1,
    const float* __restrict__ gg1, const float* __restrict__ gb1,
    const float* __restrict__ x3, const float* __restrict__ acc3,
    const float* __restrict__ g3, const float* __restrict__ b3,
    float* __restrict__ out, float* __restrict__ acc) {
  int idx = blockIdx.x * 256 + threadIdx.x;
  int c = (idx / HW) % CC;
  int b = idx / (CC * HW);
  int grp = c >> 2;
  const float invNg = 1.f / (4.f * HW);
  // group norm 0
  float m0 = accG0[(b * 16 + grp) * 2] * invNg;
  float v0 = accG0[(b * 16 + grp) * 2 + 1] * invNg - m0 * m0;
  float s0 = rsqrtf(v0 + EPSV) * gg0[c], h0 = gb0[c] - m0 * s0;
  // group norm 1
  float m1 = accG1[(b * 16 + grp) * 2] * invNg;
  float v1 = accG1[(b * 16 + grp) * 2 + 1] * invNg - m1 * m1;
  float s1 = rsqrtf(v1 + EPSV) * gg1[c], h1 = gb1[c] - m1 * s1;
  // branch-3 BN
  float s3, h3;
  bn_affine(acc3, g3, b3, c, s3, h3);
  float v = x[idx] + fmaxf(ds0[idx] * s0 + h0, 0.f) +
            fmaxf(ds1[idx] * s1 + h1, 0.f) + fmaxf(x3[idx] * s3 + h3, 0.f);
  out[idx] = v;
  wave_acc(v, acc, c);
}

// ---------------------------------- final: gelu(x + relu(bn(x0))), exact gelu
__global__ __launch_bounds__(256) void k_final(
    const float* __restrict__ x, const float* __restrict__ x0,
    const float* __restrict__ acc, const float* __restrict__ g,
    const float* __restrict__ b, float* __restrict__ out) {
  int idx = blockIdx.x * 256 + threadIdx.x;
  int c = (idx / HW) % CC;
  float sc, sh;
  bn_affine(acc, g, b, c, sc, sh);
  float t = fmaxf(x0[idx] * sc + sh, 0.f);
  float v = x[idx] + t;
  out[idx] = 0.5f * v * (1.f + erff(v * 0.70710678118654752f));
}

// =============================================================== host launch
extern "C" void kernel_launch(void* const* d_in, const int* in_sizes, int n_in,
                              void* d_out, int out_size, void* d_ws, size_t ws_size,
                              hipStream_t stream) {
  const float* x     = (const float*)d_in[0];
  const float* w11   = (const float*)d_in[1];
  const float* w12   = (const float*)d_in[2];
  const float* w21   = (const float*)d_in[3];
  const float* w22   = (const float*)d_in[4];
  const float* w31   = (const float*)d_in[5];
  const float* w32   = (const float*)d_in[6];
  const float* w33   = (const float*)d_in[7];
  const float* cm_g  = (const float*)d_in[8];
  const float* cm_b  = (const float*)d_in[9];
  const float* off_w = (const float*)d_in[10];
  const float* off_b = (const float*)d_in[11];
  const float* off_g = (const float*)d_in[12];
  const float* off_be= (const float*)d_in[13];
  const float* ds_w  = (const float*)d_in[14];
  const float* ds_b  = (const float*)d_in[15];
  const float* gn_g  = (const float*)d_in[16];
  const float* gn_b  = (const float*)d_in[17];
  const float* bn_g  = (const float*)d_in[18];
  const float* bn_b  = (const float*)d_in[19];

  float* ws = (float*)d_ws;
  const size_t F = (size_t)FSZ;
  float* A   = ws;
  float* Bb  = ws + F;
  float* DS0 = ws + 2 * F;
  float* DS1 = ws + 3 * F;
  float* X0  = ws + 4 * F;
  float* OFF = ws + 5 * F;                          // 2*22*HW raw offsets
  unsigned short* WBF = (unsigned short*)(OFF + (size_t)BB * 2 * KK * HW);
  float* ACC = OFF + (size_t)BB * 2 * KK * HW + (2 * CC * LDIM) / 2;

  const int NB_F   = FSZ / 256;                  // 8192
  const int NB_OFF = (BB * 2 * KK * HW) / 256;   // 2816
  const int NB_DS  = (BB * HW) / 16;             // 2048

  hipMemsetAsync(ACC, 0, 2048 * sizeof(float), stream);
  k_f32_to_bf16<<<(2 * CC * LDIM + 255) / 256, 256, 0, stream>>>(
      ds_w, WBF, 2 * CC * LDIM);

  float* accC[7];
  for (int s = 0; s < 7; ++s) accC[s] = ACC + s * 128;
  float* accO0 = ACC + 896, *accO1 = ACC + 960;
  float* accG0 = ACC + 1024, *accG1 = ACC + 1088;
  float* accFin = ACC + 1152;

  // ---- branch 1: (1x15) -> (1x13) -> dsconv morph 0
  k_dwconv<<<NB_F, 256, 0, stream>>>(x, nullptr, nullptr, nullptr, w11, A, accC[0], 1, 15, 0);
  k_dwconv<<<NB_F, 256, 0, stream>>>(A, accC[0], cm_g + 0 * CC, cm_b + 0 * CC, w12, Bb, accC[1], 1, 13, 1);
  k_offconv<<<NB_OFF, 256, 0, stream>>>(Bb, accC[1], cm_g + 1 * CC, cm_b + 1 * CC, off_w, off_b, OFF, accO0);
  k_dsconv_wmma<<<NB_DS, 32, 0, stream>>>(Bb, accC[1], cm_g + 1 * CC, cm_b + 1 * CC,
                                          OFF, accO0, off_g, off_be,
                                          WBF, ds_b, DS0, accG0, 0);

  // ---- branch 2: (15x1) -> (13x1) -> dsconv morph 1
  k_dwconv<<<NB_F, 256, 0, stream>>>(x, nullptr, nullptr, nullptr, w21, A, accC[2], 15, 1, 0);
  k_dwconv<<<NB_F, 256, 0, stream>>>(A, accC[2], cm_g + 2 * CC, cm_b + 2 * CC, w22, Bb, accC[3], 13, 1, 1);
  k_offconv<<<NB_OFF, 256, 0, stream>>>(Bb, accC[3], cm_g + 3 * CC, cm_b + 3 * CC,
                                        off_w + 2 * KK * CC * 9, off_b + 2 * KK, OFF, accO1);
  k_dsconv_wmma<<<NB_DS, 32, 0, stream>>>(Bb, accC[3], cm_g + 3 * CC, cm_b + 3 * CC,
                                          OFF, accO1, off_g + 2 * KK, off_be + 2 * KK,
                                          WBF + CC * LDIM, ds_b + CC, DS1, accG1, 1);

  // ---- branch 3: (15x15) -> (13x13) -> (11x11); raw result stays in A
  k_dwconv<<<NB_F, 256, 0, stream>>>(x, nullptr, nullptr, nullptr, w31, A, accC[4], 15, 15, 0);
  k_dwconv<<<NB_F, 256, 0, stream>>>(A, accC[4], cm_g + 4 * CC, cm_b + 4 * CC, w32, Bb, accC[5], 13, 13, 1);
  k_dwconv<<<NB_F, 256, 0, stream>>>(Bb, accC[5], cm_g + 5 * CC, cm_b + 5 * CC, w33, A, accC[6], 11, 11, 1);

  // ---- fuse all branches (GN + BN applied inline) then final BN + GELU
  k_sum4_reduce<<<NB_F, 256, 0, stream>>>(x,
                                          DS0, accG0, gn_g, gn_b,
                                          DS1, accG1, gn_g + CC, gn_b + CC,
                                          A, accC[6], cm_g + 6 * CC, cm_b + 6 * CC,
                                          X0, accFin);
  k_final<<<NB_F, 256, 0, stream>>>(x, X0, accFin, bn_g, bn_b, (float*)d_out);
}
